// MultiHeadedAttention_75746043232850
// MI455X (gfx1250) — matrix operand order repaired and checked by reading, not compiled, not run on hardware
//
#include <hip/hip_runtime.h>

typedef __attribute__((ext_vector_type(16))) _Float16 v16h;
typedef __attribute__((ext_vector_type(8)))  _Float16 v8h;
typedef __attribute__((ext_vector_type(8)))  float    v8f;
typedef __attribute__((ext_vector_type(4)))  int      v4i;

constexpr int Bc  = 2;
constexpr int Sc  = 2048;
constexpr int Dc  = 1024;
constexpr int Hc  = 16;
constexpr int DKc = 64;
constexpr int Mc  = Bc * Sc;   // 4096 rows
constexpr int LDK = 72;        // padded LDS row stride in halves (144B: bank-spread)

static __device__ __forceinline__ v16h cat8(v8h a, v8h b) {
  return __builtin_shufflevector(a, b, 0,1,2,3,4,5,6,7,8,9,10,11,12,13,14,15);
}

static __device__ __forceinline__ v8f wmma_f16(v16h a, v16h b, v8f c) {
  // D = A(16x32 f16) * B(32x16 f16) + C(16x16 f32)
  return __builtin_amdgcn_wmma_f32_16x16x32_f16(false, a, false, b, (short)0, c,
                                                false, false);
}

// ---------------------------------------------------------------------------
// Async global -> LDS copy (CDNA5 GLOBAL_LOAD_ASYNC_TO_LDS_B128, ASYNCcnt),
// guarded with a synchronous fallback so compilation never breaks.
// Builtin signature (from toolchain diagnostics): 
//   (v4i addrspace(1)*, v4i addrspace(3)*, imm int offset, imm int cpol)
// ---------------------------------------------------------------------------
#if defined(__AMDGCN__) &&                                                   \
    __has_builtin(__builtin_amdgcn_global_load_async_to_lds_b128) &&         \
    __has_builtin(__builtin_amdgcn_s_wait_asynccnt)
#define ASYNC_COPY 1
#else
#define ASYNC_COPY 0
#endif

typedef __attribute__((address_space(1))) v4i* gbl_v4i_p;
typedef __attribute__((address_space(3))) v4i* lds_v4i_p;

static __device__ __forceinline__ void copy16B(_Float16* lds_dst,
                                               const _Float16* gsrc) {
#if ASYNC_COPY
  // Flat LDS address carries the LDS byte offset in its low 32 bits.
  __builtin_amdgcn_global_load_async_to_lds_b128(
      (gbl_v4i_p)(unsigned long long)gsrc,
      (lds_v4i_p)(unsigned int)(unsigned long long)lds_dst, 0, 0);
#else
  *(v8h*)lds_dst = *(const v8h*)gsrc;
#endif
}

template <int N>
static __device__ __forceinline__ void wait_async() {
#if ASYNC_COPY
  __builtin_amdgcn_s_wait_asynccnt(N);
#endif
}

// ---------------------------------------------------------------------------
// K0: fp32 -> f16 conversion
// ---------------------------------------------------------------------------
__global__ void cvt_f16_kernel(const float* __restrict__ src,
                               _Float16* __restrict__ dst, int n) {
  int i = blockIdx.x * blockDim.x + threadIdx.x;
  int stride = gridDim.x * blockDim.x;
  for (; i < n; i += stride) dst[i] = (_Float16)src[i];
}

// ---------------------------------------------------------------------------
// K1/K3: C[m,n] = sum_k A[m,k] * W[n,k] + bias[n]   (M=4096, N=K=1024)
//   MODE 0: f16 out row-major [M,N]            (Q, K projections)
//   MODE 1: f16 out[(b*N + n)*S + s], b=m/S    (V projection, per-head transpose)
//   MODE 2: f32 out row-major [M,N]            (output projection)
// Block: 256 threads (8 waves as 4x2), block tile 128(M) x 128(N),
// each wave computes a 32x64 tile = 2x4 WMMA tiles (8 WMMAs / k-step).
// Weights are L2-resident (2MB f16 vs 192MB L2): direct global fragments.
// ---------------------------------------------------------------------------
template <int MODE>
__global__ __launch_bounds__(256) void gemm_wmma_kernel(
    const _Float16* __restrict__ A, const _Float16* __restrict__ W,
    const float* __restrict__ bias, void* __restrict__ out) {
  const int lane = threadIdx.x & 31;
  const int wave = threadIdx.x >> 5;
  const int lm = lane & 15;       // row/col within a 16-tile
  const int lh = lane >> 4;       // half-wave select

  const int m0 = blockIdx.x * 128 + (wave >> 1) * 32;
  const int n0 = blockIdx.y * 128 + (wave & 1) * 64;

  const _Float16* arow0 = A + (size_t)(m0 + lm) * Dc;
  const _Float16* arow1 = arow0 + (size_t)16 * Dc;
  const _Float16* wrow[4];
#pragma unroll
  for (int t = 0; t < 4; ++t) wrow[t] = W + (size_t)(n0 + t * 16 + lm) * Dc;

  v8f acc[2][4] = {};

  for (int k0 = 0; k0 < Dc; k0 += 32) {
    if (k0 + 512 < Dc) {   // prefetch streamed activation rows (global_prefetch)
      __builtin_prefetch(arow0 + k0 + 512, 0, 1);
      __builtin_prefetch(arow1 + k0 + 512, 0, 1);
    }
    // A fragments (16x32): halves 0..7 at k0+lh*8, halves 8..15 at k0+16+lh*8
    v16h a0 = cat8(*(const v8h*)(arow0 + k0 + lh * 8),
                   *(const v8h*)(arow0 + k0 + 16 + lh * 8));
    v16h a1 = cat8(*(const v8h*)(arow1 + k0 + lh * 8),
                   *(const v8h*)(arow1 + k0 + 16 + lh * 8));
    // B fragments (32x16): lane = column n (row of W), halves 0..7 at k0+lh*8, 8..15 at k0+16+lh*8
    v16h bfr[4];
#pragma unroll
    for (int t = 0; t < 4; ++t)
      bfr[t] = cat8(*(const v8h*)(wrow[t] + k0 + lh * 8),
                    *(const v8h*)(wrow[t] + k0 + 16 + lh * 8));
#pragma unroll
    for (int t = 0; t < 4; ++t) {
      acc[0][t] = wmma_f16(a0, bfr[t], acc[0][t]);
      acc[1][t] = wmma_f16(a1, bfr[t], acc[1][t]);
    }
  }

  float bn[4];
#pragma unroll
  for (int t = 0; t < 4; ++t) bn[t] = bias[n0 + t * 16 + lm];

#pragma unroll
  for (int mt = 0; mt < 2; ++mt) {
#pragma unroll
    for (int nt = 0; nt < 4; ++nt) {
      const int ncol = n0 + nt * 16 + lm;
#pragma unroll
      for (int r = 0; r < 8; ++r) {
        const int mrow = m0 + mt * 16 + r + lh * 8;   // C layout: m = r + lh*8
        const float v = acc[mt][nt][r] + bn[nt];
        if (MODE == 2) {
          ((float*)out)[(size_t)mrow * Dc + ncol] = v;
        } else if (MODE == 0) {
          ((_Float16*)out)[(size_t)mrow * Dc + ncol] = (_Float16)v;
        } else {
          const int b = mrow >> 11, s = mrow & (Sc - 1);
          ((_Float16*)out)[((size_t)b * Dc + ncol) * Sc + s] = (_Float16)v;
        }
      }
    }
  }
}

// ---------------------------------------------------------------------------
// K2: flash attention. One (batch, head) per blockIdx.y, 128 query rows per
// block (8 waves x 16 rows). kv loop in 64-wide blocks, K/V tiles staged in
// LDS with double-buffered async global->LDS copies (all 8 waves share them).
//   Qh, Kh: f16 [B*S, D] row-major (head h = cols h*64..h*64+63)
//   Vt:     f16 [B, D, S] (per-batch transposed V)
//   ctx:    f16 [B*S, D] row-major
// ---------------------------------------------------------------------------
__global__ __launch_bounds__(256) void attn_wmma_kernel(
    const _Float16* __restrict__ Qh, const _Float16* __restrict__ Kh,
    const _Float16* __restrict__ Vt, _Float16* __restrict__ ctx) {
  __shared__ _Float16 k_tile[2][64 * LDK];   // [kv][d], padded rows
  __shared__ _Float16 v_tile[2][64 * LDK];   // [d][kv], padded rows
  __shared__ _Float16 p_tile[8][16 * LDK];   // per-wave P tile

  const int tid  = threadIdx.x;
  const int lane = tid & 31;
  const int wave = tid >> 5;
  const int lm = lane & 15;
  const int lh = lane >> 4;

  const int bh = blockIdx.y;                // 0..31
  const int b = bh >> 4, h = bh & 15;
  const int q0 = blockIdx.x * 128 + wave * 16;

  // Q fragments for d=0..31 and d=32..63 (loaded once, direct global)
  const _Float16* qrow = Qh + (size_t)(b * Sc + q0 + lm) * Dc + h * DKc;
  const v16h qa0 = cat8(*(const v8h*)(qrow + lh * 8),
                        *(const v8h*)(qrow + 16 + lh * 8));
  const v16h qa1 = cat8(*(const v8h*)(qrow + 32 + lh * 8),
                        *(const v8h*)(qrow + 48 + lh * 8));

  const _Float16* kg = Kh + (size_t)(b * Sc) * Dc + h * DKc;
  const _Float16* vg = Vt + (size_t)(b * Dc + h * DKc) * Sc;

  // Cooperative tile copy: 64 rows x 128B each for K and V = 512 16B chunks
  // per tile; 256 threads x 2 chunks. 4 async instructions per wave per stage.
  auto issue = [&](int buf, int kv0) {
#pragma unroll
    for (int i = 0; i < 2; ++i) {
      const int c = tid + i * 256;
      const int row = c >> 3;
      const int off = (c & 7) * 8;
      copy16B(&k_tile[buf][row * LDK + off],
              kg + (size_t)(kv0 + row) * Dc + off);
      copy16B(&v_tile[buf][row * LDK + off],
              vg + (size_t)row * Sc + kv0 + off);
    }
  };

  v8f o[4] = {};
  float mrow[8], lrow[8];
#pragma unroll
  for (int r = 0; r < 8; ++r) { mrow[r] = -1e30f; lrow[r] = 0.0f; }

  _Float16* pl = p_tile[wave];
  constexpr int NB = Sc / 64;   // 32 kv blocks

  issue(0, 0);
  for (int kb = 0; kb < NB; ++kb) {
    const int cur = kb & 1;
    if (kb + 1 < NB) {
      issue(cur ^ 1, (kb + 1) * 64);
      wait_async<4>();          // 4 newest still in flight -> current tile done
    } else {
      wait_async<0>();
    }
    __syncthreads();

    const _Float16* kl = k_tile[cur];
    const _Float16* vl = v_tile[cur];

    // ----- scores: four 16x16 kv tiles, contraction over d (2 WMMAs each)
    v8f s[4] = {};
#pragma unroll
    for (int t = 0; t < 4; ++t) {
      const _Float16* kr = kl + (t * 16 + lm) * LDK;
      const v16h kb0 = cat8(*(const v8h*)(kr + lh * 8),
                            *(const v8h*)(kr + 16 + lh * 8));
      const v16h kb1 = cat8(*(const v8h*)(kr + 32 + lh * 8),
                            *(const v8h*)(kr + 48 + lh * 8));
      s[t] = wmma_f16(qa0, kb0, s[t]);
      s[t] = wmma_f16(qa1, kb1, s[t]);
    }

    // ----- online softmax (row = r + lh*8, values spread over 16 lanes)
    const float scl = 0.125f;   // 1/sqrt(64)
#pragma unroll
    for (int r = 0; r < 8; ++r) {
      const float x0 = s[0][r] * scl, x1 = s[1][r] * scl;
      const float x2 = s[2][r] * scl, x3 = s[3][r] * scl;
      float mx = fmaxf(fmaxf(x0, x1), fmaxf(x2, x3));
      mx = fmaxf(mx, __shfl_xor(mx, 1));
      mx = fmaxf(mx, __shfl_xor(mx, 2));
      mx = fmaxf(mx, __shfl_xor(mx, 4));
      mx = fmaxf(mx, __shfl_xor(mx, 8));
      const float mnew = fmaxf(mrow[r], mx);
      const float corr = __expf(mrow[r] - mnew);
      const float p0 = __expf(x0 - mnew), p1 = __expf(x1 - mnew);
      const float p2 = __expf(x2 - mnew), p3 = __expf(x3 - mnew);
      float sum = (p0 + p1) + (p2 + p3);
      sum += __shfl_xor(sum, 1);
      sum += __shfl_xor(sum, 2);
      sum += __shfl_xor(sum, 4);
      sum += __shfl_xor(sum, 8);
      lrow[r] = lrow[r] * corr + sum;
      mrow[r] = mnew;
      o[0][r] *= corr; o[1][r] *= corr; o[2][r] *= corr; o[3][r] *= corr;
      _Float16* prw = pl + (r + lh * 8) * LDK;
      prw[lm]      = (_Float16)p0;
      prw[16 + lm] = (_Float16)p1;
      prw[32 + lm] = (_Float16)p2;
      prw[48 + lm] = (_Float16)p3;
    }

    // ----- re-layout P (C-layout -> two A fragments) through per-wave LDS
    const _Float16* prd = pl + lm * LDK;
    const v16h pa0 = cat8(*(const v8h*)(prd + lh * 8),
                          *(const v8h*)(prd + 16 + lh * 8));
    const v16h pa1 = cat8(*(const v8h*)(prd + 32 + lh * 8),
                          *(const v8h*)(prd + 48 + lh * 8));

    // ----- O += P @ V  (4 d-tiles, contraction over 64 kv = 2 WMMAs each)
#pragma unroll
    for (int t = 0; t < 4; ++t) {
      const _Float16* vr = vl + (t * 16 + lm) * LDK;
      const v16h vb0 = cat8(*(const v8h*)(vr + lh * 8),
                            *(const v8h*)(vr + 16 + lh * 8));
      const v16h vb1 = cat8(*(const v8h*)(vr + 32 + lh * 8),
                            *(const v8h*)(vr + 48 + lh * 8));
      o[t] = wmma_f16(pa0, vb0, o[t]);
      o[t] = wmma_f16(pa1, vb1, o[t]);
    }
    __syncthreads();   // everyone done with buffer `cur` before it is refilled
  }

  // ----- normalize and store context (f16 row-major [B*S, D])
#pragma unroll
  for (int t = 0; t < 4; ++t) {
#pragma unroll
    for (int r = 0; r < 8; ++r) {
      const int m = q0 + r + lh * 8;
      ctx[(size_t)(b * Sc + m) * Dc + h * DKc + t * 16 + lm] =
          (_Float16)(o[t][r] / lrow[r]);
    }
  }
}

// ---------------------------------------------------------------------------
// Host launcher
// ---------------------------------------------------------------------------
extern "C" void kernel_launch(void* const* d_in, const int* in_sizes, int n_in,
                              void* d_out, int out_size, void* d_ws,
                              size_t ws_size, hipStream_t stream) {
  (void)in_sizes; (void)n_in; (void)out_size; (void)ws_size;

  const float* x  = (const float*)d_in[0];
  const float* Wq = (const float*)d_in[1];
  const float* bq = (const float*)d_in[2];
  const float* Wk = (const float*)d_in[3];
  const float* bk = (const float*)d_in[4];
  const float* Wv = (const float*)d_in[5];
  const float* bv = (const float*)d_in[6];
  const float* Wo = (const float*)d_in[7];
  const float* bo = (const float*)d_in[8];
  float* out = (float*)d_out;

  char* ws = (char*)d_ws;
  size_t off = 0;
  auto alloc = [&](size_t bytes) -> void* {
    void* p = ws + off;
    off = (off + bytes + 255) & ~(size_t)255;
    return p;
  };

  const size_t nX = (size_t)Mc * Dc;        // 4096*1024
  const size_t nW = (size_t)Dc * Dc;        // 1024*1024

  _Float16* xh  = (_Float16*)alloc(nX * 2);
  _Float16* wqh = (_Float16*)alloc(nW * 2);
  _Float16* wkh = (_Float16*)alloc(nW * 2);
  _Float16* wvh = (_Float16*)alloc(nW * 2);
  _Float16* woh = (_Float16*)alloc(nW * 2);
  _Float16* Qh  = (_Float16*)alloc(nX * 2);
  _Float16* Kh  = (_Float16*)alloc(nX * 2);
  _Float16* Vt  = (_Float16*)alloc(nX * 2);
  _Float16* ctx = xh;   // reuse: xh dead after the V projection (stream-ordered)

  // --- convert inputs to f16
  cvt_f16_kernel<<<2048, 256, 0, stream>>>(x,  xh,  (int)nX);
  cvt_f16_kernel<<<1024, 256, 0, stream>>>(Wq, wqh, (int)nW);
  cvt_f16_kernel<<<1024, 256, 0, stream>>>(Wk, wkh, (int)nW);
  cvt_f16_kernel<<<1024, 256, 0, stream>>>(Wv, wvh, (int)nW);
  cvt_f16_kernel<<<1024, 256, 0, stream>>>(Wo, woh, (int)nW);

  // --- QKV projections (block tile 128x128)
  dim3 gg(Mc / 128, Dc / 128);   // 32 x 8
  gemm_wmma_kernel<0><<<gg, 256, 0, stream>>>(xh, wqh, bq, Qh);
  gemm_wmma_kernel<0><<<gg, 256, 0, stream>>>(xh, wkh, bk, Kh);
  gemm_wmma_kernel<1><<<gg, 256, 0, stream>>>(xh, wvh, bv, Vt);

  // --- attention
  attn_wmma_kernel<<<dim3(Sc / 128, Bc * Hc), 256, 0, stream>>>(Qh, Kh, Vt, ctx);

  // --- output projection (fp32 result)
  gemm_wmma_kernel<2><<<gg, 256, 0, stream>>>(ctx, woh, bo, out);
}
